// PlyMoE_53515292508315
// MI455X (gfx1250) — matrix operand order repaired and checked by reading, not compiled, not run on hardware
//
#include <hip/hip_runtime.h>

typedef __attribute__((ext_vector_type(16))) _Float16 v16h;
typedef __attribute__((ext_vector_type(8)))  _Float16 v8h;
typedef __attribute__((ext_vector_type(8)))  float    v8f;

namespace {
constexpr int NEXP = 8;
constexpr int KDIM = 1024;
constexpr int NDIM = 4096;
constexpr int MDIM = 16384;

constexpr int BM = 128;
constexpr int BN = 128;
constexpr int BK = 32;
constexpr int NKSTEP = KDIM / BK;           // 32 k-steps
constexpr int LDA = BK + 8;                 // 40 halves = 80B row stride (16B aligned, pad vs bank conflicts)
constexpr int LDB = BK + 8;
constexpr int SORT_BLK = 256;
constexpr int NSORT_BLK = MDIM / SORT_BLK;  // 64
constexpr int MAX_TILES = MDIM / BM + NEXP + 8;  // 144 (>= 128 + 8 boundary tiles)
}

// ---------------- Phase 1a: per-block expert histogram ----------------
__global__ void moe_hist_kernel(const int* __restrict__ idx, int* __restrict__ hist) {
    __shared__ int sH[NEXP];
    const int t = threadIdx.x, b = blockIdx.x;
    if (t < NEXP) sH[t] = 0;
    __syncthreads();
    atomicAdd(&sH[idx[b * SORT_BLK + t]], 1);
    __syncthreads();
    if (t < NEXP) hist[b * NEXP + t] = sH[t];
}

// ---------------- Phase 1b: offsets + tile work list ----------------
__global__ void moe_offsets_kernel(const int* __restrict__ hist,
                                   int* __restrict__ blockOff,
                                   int* __restrict__ expertBase,
                                   int* __restrict__ tiles,
                                   int* __restrict__ numTiles) {
    __shared__ int h[NSORT_BLK * NEXP];
    __shared__ int tot[NEXP];
    __shared__ int base[NEXP + 1];
    const int t = threadIdx.x;
    for (int i = t; i < NSORT_BLK * NEXP; i += blockDim.x) h[i] = hist[i];
    __syncthreads();
    if (t < NEXP) {
        int s = 0;
        for (int b = 0; b < NSORT_BLK; ++b) s += h[b * NEXP + t];
        tot[t] = s;
    }
    __syncthreads();
    if (t == 0) {
        base[0] = 0;
        for (int e = 0; e < NEXP; ++e) base[e + 1] = base[e] + tot[e];
    }
    __syncthreads();
    if (t < NEXP) {
        int run = base[t];
        for (int b = 0; b < NSORT_BLK; ++b) {
            blockOff[b * NEXP + t] = run;
            run += h[b * NEXP + t];
        }
    }
    if (t < NEXP + 1) expertBase[t] = base[t];
    if (t == 0) {
        int nt = 0;
        for (int e = 0; e < NEXP; ++e) {
            const int s = base[e], en = base[e + 1];
            if (s == en) continue;
            const int t0 = s / BM, t1 = (en + BM - 1) / BM;
            for (int ti = t0; ti < t1; ++ti) {
                const int rb = ti * BM;
                tiles[nt * 4 + 0] = e;
                tiles[nt * 4 + 1] = rb;
                tiles[nt * 4 + 2] = (s  > rb)      ? s  : rb;        // valid start
                tiles[nt * 4 + 3] = (en < rb + BM) ? en : rb + BM;   // valid end
                ++nt;
            }
        }
        *numTiles = nt;
    }
}

// ---------------- Phase 1c: stable rank + scatter gather-map ----------------
__global__ void moe_scatter_kernel(const int* __restrict__ idx,
                                   const int* __restrict__ blockOff,
                                   int* __restrict__ srcMap) {
    __shared__ int sE[SORT_BLK];
    const int t = threadIdx.x, b = blockIdx.x;
    const int tok = b * SORT_BLK + t;
    const int e = idx[tok];
    sE[t] = e;
    __syncthreads();
    int r = 0;
    for (int j = 0; j < t; ++j) r += (sE[j] == e);   // stable within block; blocks ordered
    srcMap[blockOff[b * NEXP + e] + r] = tok;
}

// ---------------- Phase 2: grouped WMMA GEMM ----------------
// A tile: gathered copy -> async global->LDS (ASYNCcnt), 4-deep LDS ring.
// B tile: global->reg, v_perm transpose, ds_store -> double-buffered LDS.
// k-loop unrolled x4 so all ring/buffer indices are compile-time constants.
__global__ __launch_bounds__(256, 1)
void moe_gemm_kernel(const _Float16* __restrict__ xg,
                     const _Float16* __restrict__ wg,
                     const int* __restrict__ tiles,
                     const int* __restrict__ numTiles,
                     const int* __restrict__ srcMap,
                     _Float16* __restrict__ outg) {
    __shared__ _Float16 As[4][BM * LDA];   // 4 x 10KB async ring
    __shared__ _Float16 Bs[2][BN * LDB];   // 2 x 10KB
    __shared__ int sSrc[BM];

    const int tileId = blockIdx.x;
    if (tileId >= numTiles[0]) return;

    const int e       = tiles[tileId * 4 + 0];
    const int rowBase = tiles[tileId * 4 + 1];
    const int vs      = tiles[tileId * 4 + 2];
    const int ve      = tiles[tileId * 4 + 3];
    const int nBase   = blockIdx.y * BN;
    const int t       = threadIdx.x;

    // Invalid rows clamp to token 0: valid memory, garbage confined to their
    // own output row, which is masked at store.
    if (t < BM) {
        const int r = rowBase + t;
        sSrc[t] = (r >= vs && r < ve) ? srcMap[r] : 0;
    }
    __syncthreads();

    const _Float16* we = wg + (size_t)e * KDIM * NDIM;

    // per-thread staging mapping:
    //   A: two 16B chunks, rows (t>>2) and (t>>2)+64, k-segment (t&3)*8
    //   B: two n-contiguous 16B rows at k and k+1 -> transposed into LDS as packed b32
    const int rowA = t >> 2;
    const int segA = (t & 3) * 8;
    const int kB   = (t >> 4) * 2;
    const int nB   = (t & 15) * 8;
    const int s0 = sSrc[rowA];
    const int s1 = sSrc[rowA + 64];

    // LDS byte addresses (low 32 bits of flat LDS pointer == LDS offset)
    const unsigned ldsA0 = (unsigned)(uintptr_t)&As[0][rowA * LDA + segA];
    const unsigned ldsA1 = (unsigned)(uintptr_t)&As[0][(rowA + 64) * LDA + segA];
    constexpr unsigned ABUF_BYTES = (unsigned)(BM * LDA * sizeof(_Float16)); // 10240
    const unsigned long long xBase = (unsigned long long)(uintptr_t)xg;
    const unsigned gaOff0 = (unsigned)(((unsigned)s0 * KDIM + segA) * sizeof(_Float16));
    const unsigned gaOff1 = (unsigned)(((unsigned)s1 * KDIM + segA) * sizeof(_Float16));

    // async gathered copy of one A k-slab into ring buffer `buf` (compile-time buf)
    auto asyncA = [&](int kb, int buf) {
        const unsigned off = (unsigned)(kb * BK * sizeof(_Float16));
        const unsigned bufOff = (unsigned)buf * ABUF_BYTES;
        asm volatile("global_load_async_to_lds_b128 %0, %1, %2"
                     :: "v"(ldsA0 + bufOff), "v"(gaOff0 + off), "s"(xBase) : "memory");
        asm volatile("global_load_async_to_lds_b128 %0, %1, %2"
                     :: "v"(ldsA1 + bufOff), "v"(gaOff1 + off), "s"(xBase) : "memory");
    };

    const _Float16* pb = we + (size_t)kB * NDIM + nBase + nB;
    union V8U4 { v8h v; unsigned int u[4]; };
    V8U4 rb0, rb1;

    auto gloadB = [&](int kb) {
        const _Float16* p = pb + (size_t)(kb * BK) * NDIM;
        rb0.v = *(const v8h*)p;
        rb1.v = *(const v8h*)(p + NDIM);
    };

    auto lstoreB = [&](int buf) {
        #pragma unroll
        for (int j = 0; j < 8; ++j) {
            // Bs[n][k] <- {B[k][n], B[k+1][n]} : one v_perm_b32 per packed dword
            const unsigned int sel = (j & 1) ? 0x07060302u : 0x05040100u;
            const unsigned int pk =
                __builtin_amdgcn_perm(rb1.u[j >> 1], rb0.u[j >> 1], sel);
            *(unsigned int*)&Bs[buf][(nB + j) * LDB + kB] = pk;
        }
    };

    const int lane   = t & 31;
    const int wid    = t >> 5;
    const int wm     = wid >> 2;   // 0..1 : 64 rows per wave
    const int wn     = wid & 3;    // 0..3 : 32 cols per wave
    const int laneLo = lane & 15;
    const int laneHi = lane >> 4;

    v8f acc[4][2] = {};

    auto compute = [&](int abuf, int bbuf) {
        const _Float16* A = As[abuf];
        const _Float16* B = Bs[bbuf];
        // Load ALL fragments first: 12 independent ds_load_b128, so the
        // scheduler can stagger partial s_wait_dscnt against the WMMAs.
        v16h bfrag[2], afrag[4];
        #pragma unroll
        for (int nf = 0; nf < 2; ++nf) {
            // B 32x16 layout: lane n%16, lanes 0-15 hold K 0-7 & 16-23, lanes 16-31 hold K 8-15 & 24-31
            const _Float16* p = B + (wn * 32 + nf * 16 + laneLo) * LDB;
            v8h lo = *(const v8h*)(p + laneHi * 8);
            v8h hi = *(const v8h*)(p + 16 + laneHi * 8);
            bfrag[nf] = __builtin_shufflevector(lo, hi, 0,1,2,3,4,5,6,7,8,9,10,11,12,13,14,15);
        }
        #pragma unroll
        for (int mf = 0; mf < 4; ++mf) {
            // A 16x32 layout: lanes 0-15 hold K 0-7 & 16-23, lanes 16-31 hold K 8-15 & 24-31
            const _Float16* p = A + (wm * 64 + mf * 16 + laneLo) * LDA;
            v8h lo = *(const v8h*)(p + laneHi * 8);
            v8h hi = *(const v8h*)(p + 16 + laneHi * 8);
            afrag[mf] = __builtin_shufflevector(lo, hi, 0,1,2,3,4,5,6,7,8,9,10,11,12,13,14,15);
        }
        #pragma unroll
        for (int mf = 0; mf < 4; ++mf) {
            #pragma unroll
            for (int nf = 0; nf < 2; ++nf) {
                acc[mf][nf] = __builtin_amdgcn_wmma_f32_16x16x32_f16(
                    false, afrag[mf], false, bfrag[nf], (short)0, acc[mf][nf], false, false);
            }
        }
    };

    // ---- software pipeline ----
    // Invariants entering iteration kb:
    //   A(kb) resident, A(kb+1) async in flight (in-order => asynccnt<=2 proves A(kb+1))
    //   B(kb) in LDS, B(kb+1) global loads in flight (issued one compute phase earlier)
    asyncA(0, 0);
    asyncA(1, 1);
    gloadB(0);
    lstoreB(0);
    asm volatile("s_wait_asynccnt 2" ::: "memory");   // A(0) resident
    __syncthreads();
    gloadB(1);                                        // in flight during compute(0)

    #pragma unroll 4
    for (int kb = 0; kb < NKSTEP; ++kb) {
        compute(kb & 3, kb & 1);
        if (kb + 1 < NKSTEP) {
            __syncthreads();                          // readers of kb-1 buffers done
            if (kb + 2 < NKSTEP) {
                asyncA(kb + 2, (kb + 2) & 3);         // ring slot free since step kb-2
                lstoreB((kb + 1) & 1);                // regs loaded a full phase ago
                asm volatile("s_wait_asynccnt 2" ::: "memory");  // A(kb+1) resident
            } else {
                lstoreB((kb + 1) & 1);
                asm volatile("s_wait_asynccnt 0" ::: "memory");
            }
            __syncthreads();                          // publish kb+1 buffers (dscnt only)
            if (kb + 2 < NKSTEP) gloadB(kb + 2);      // in flight during compute(kb+1)
        }
    }

    // epilogue: f32 -> f16; fast path for interior (fully-valid) tiles
    const bool fullTile = (vs == rowBase) && (ve == rowBase + BM);
    if (fullTile) {
        #pragma unroll
        for (int mf = 0; mf < 4; ++mf) {
            #pragma unroll
            for (int nf = 0; nf < 2; ++nf) {
                const int col  = nBase + wn * 32 + nf * 16 + laneLo;
                const int row0 = rowBase + wm * 64 + mf * 16 + laneHi * 8;
                #pragma unroll
                for (int v = 0; v < 8; ++v)
                    outg[(size_t)(row0 + v) * NDIM + col] = (_Float16)acc[mf][nf][v];
            }
        }
    } else {
        #pragma unroll
        for (int mf = 0; mf < 4; ++mf) {
            #pragma unroll
            for (int nf = 0; nf < 2; ++nf) {
                const int col  = nBase + wn * 32 + nf * 16 + laneLo;
                const int row0 = rowBase + wm * 64 + mf * 16 + laneHi * 8;
                #pragma unroll
                for (int v = 0; v < 8; ++v) {
                    const int row = row0 + v;
                    if (row >= vs && row < ve)
                        outg[(size_t)row * NDIM + col] = (_Float16)acc[mf][nf][v];
                }
            }
        }
    }
}

extern "C" void kernel_launch(void* const* d_in, const int* in_sizes, int n_in,
                              void* d_out, int out_size, void* d_ws, size_t ws_size,
                              hipStream_t stream) {
    const _Float16* x   = (const _Float16*)d_in[0];  // [M,K] fp16
    const int*      idx = (const int*)d_in[1];       // [M] int32
    const _Float16* w   = (const _Float16*)d_in[2];  // [E,K,N] fp16
    _Float16*       out = (_Float16*)d_out;          // [M,N] fp16 (sorted-token order)

    // workspace layout (ints), ~72KB total
    int* W          = (int*)d_ws;
    int* hist       = W;                              // 64*8
    int* blockOff   = W + 512;                        // 64*8
    int* expertBase = W + 1024;                       // 9
    int* numTiles   = W + 1036;                       // 1
    int* tiles      = W + 1040;                       // MAX_TILES*4
    int* srcMap     = W + 1040 + MAX_TILES * 4;       // M

    hipLaunchKernelGGL(moe_hist_kernel, dim3(NSORT_BLK), dim3(SORT_BLK), 0, stream, idx, hist);
    hipLaunchKernelGGL(moe_offsets_kernel, dim3(1), dim3(256), 0, stream,
                       hist, blockOff, expertBase, tiles, numTiles);
    hipLaunchKernelGGL(moe_scatter_kernel, dim3(NSORT_BLK), dim3(SORT_BLK), 0, stream,
                       idx, blockOff, srcMap);
    hipLaunchKernelGGL(moe_gemm_kernel, dim3(MAX_TILES, NDIM / BN), dim3(256), 0, stream,
                       x, w, tiles, numTiles, srcMap, out);
}
